// Simple_Factored_Model_52123723104620
// MI455X (gfx1250) — matrix operand order repaired and checked by reading, not compiled, not run on hardware
//
#include <hip/hip_runtime.h>
#include <hip/hip_bf16.h>
#include <cstdint>
#include <cstddef>

// ---------------------------------------------------------------------------
// Types for CDNA5 WMMA (wave32, 16x16x32 bf16 -> f32)
// ---------------------------------------------------------------------------
typedef __attribute__((ext_vector_type(8)))  float          v8f;
typedef __attribute__((ext_vector_type(16))) __bf16         v16bf;
typedef __attribute__((ext_vector_type(16))) unsigned short u16x16;

#define NHID 1024
#define NOBJS 32768
#define R1V 10

// ---------------------------------------------------------------------------
// Device helpers
// ---------------------------------------------------------------------------
static __device__ __forceinline__ unsigned short f2bf(float f) {
  unsigned u = __float_as_uint(f);
  unsigned r = 0x7FFFu + ((u >> 16) & 1u);
  return (unsigned short)((u + r) >> 16);
}
static __device__ __forceinline__ float preluf(float x, float a) {
  return x >= 0.f ? x : a * x;
}
static __device__ __forceinline__ float sigmoidf_(float x) {
  return 1.f / (1.f + expf(-x));
}
static __device__ __forceinline__ float rng_u01(unsigned seed, unsigned idx) {
  unsigned x = seed * 0x9E3779B9u + idx * 0x85EBCA6Bu + 0x27D4EB2Fu;
  x ^= x >> 16; x *= 0x7FEB352Du;
  x ^= x >> 15; x *= 0x846CA68Bu;
  x ^= x >> 16;
  float u = (float)(x >> 8) * (1.0f / 16777216.0f);
  return fminf(fmaxf(u, 1e-6f), 1.0f - 1e-6f);
}
static __device__ __forceinline__ float gumbel_(unsigned seed, unsigned idx) {
  float u = rng_u01(seed, idx);
  return -logf(-logf(u));
}

// Fragment loads from LDS tiles, per documented CDNA5 VGPR layouts.
// A frag 16x32: lane L -> m=L%16, kbase=(L<16)?0:8; elem e: r=e/2, half=e&1;
//   k = r<4 ? kbase+2r+half : 16+kbase+2(r-4)+half
static __device__ __forceinline__ v16bf frag_a(const unsigned short As[][34],
                                               int tm, int lane) {
  union { unsigned short s[16]; u16x16 v; } u;
  int m = tm + (lane & 15);
  int kb = (lane >> 4) ? 8 : 0;
#pragma unroll
  for (int e = 0; e < 16; ++e) {
    int r = e >> 1, hf = e & 1;
    int k = (r < 4) ? (kb + 2 * r + hf) : (16 + kb + 2 * (r - 4) + hf);
    u.s[e] = As[m][k];
  }
  return __builtin_bit_cast(v16bf, u.v);
}
// B tile stored transposed: Bs[n][k].  B frag 32x16: lane L -> n=L%16,
//   kb=(L<16)?0:8; k = e<8 ? kb+e : 16+kb+(e-8)
static __device__ __forceinline__ v16bf frag_b(const unsigned short Bs[][34],
                                               int tn, int lane) {
  union { unsigned short s[16]; u16x16 v; } u;
  int n = tn + (lane & 15);
  int kb = (lane >> 4) ? 8 : 0;
#pragma unroll
  for (int e = 0; e < 16; ++e) u.s[e] = Bs[n][(e < 8) ? (kb + e) : (16 + kb + (e - 8))];
  return __builtin_bit_cast(v16bf, u.v);
}

// ---------------------------------------------------------------------------
// WMMA GEMM: C[M=32768, N=1024] = A[M,K](bf16) x B[K,1024](f32->bf16 in LDS)
//   Block tile 128x128, 8 waves (4m x 2n); each wave 2x4 WMMA tiles of 16x16
//   (2 A frags + 4 B frags -> 8 v_wmma per 32-k step).
//   EPI==0: out_bf16[m,n] = bf16(prelu(acc + bias[n]))
//   EPI==1: logit[m] += sum_n wb[n]*prelu(acc + constv[n] + pos[m]*wpos[n]
//                                         (+ o1[m]*wo1[n]))   (fused head)
// ---------------------------------------------------------------------------
template <int EPI, bool HAS_O1>
__global__ __launch_bounds__(256) void k_gemm_wmma(
    const unsigned short* __restrict__ Aptr, int lda, int Kreal, int KT,
    const float* __restrict__ B,
    const float* __restrict__ bias, unsigned short* __restrict__ outb,
    const float* __restrict__ constv, const float* __restrict__ wpos,
    const float* __restrict__ wo1, const float* __restrict__ o1v,
    const float* __restrict__ wb, const float* __restrict__ pos,
    float* __restrict__ logit, const float* __restrict__ aprm) {
  __shared__ unsigned short As[128][34];
  __shared__ unsigned short Bs[128][34];

  const int t = threadIdx.x;
  const int lane = t & 31;
  const int w = t >> 5;
  const int wm = w >> 1;      // 0..3 -> m offset wm*32
  const int wn = w & 1;       // 0..1 -> n offset wn*64
  const int blockM = blockIdx.x * 128;
  const int n0 = blockIdx.y * 128;

  v8f acc[2][4] = {};

  // loader mapping
  const int mrow = t >> 1;          // 0..127 (A)
  const int cb = (t & 1) * 16;      // 0 or 16 (A)
  const int kr = t >> 3;            // 0..31 (B row)
  const int nb = (t & 7) * 16;      // 0..112 (B col base)

  for (int kk = 0; kk < KT; kk += 32) {
    __syncthreads();
    // ---- load A tile (128 x 32), bf16 source, unguarded vector loads ----
    {
      size_t grow = (size_t)(blockM + mrow);
      const unsigned short* Ag = Aptr + grow * (size_t)lda + kk + cb;
      uint4 p0 = *(const uint4*)(Ag);
      uint4 p1 = *(const uint4*)(Ag + 8);
      unsigned* d = (unsigned*)&As[mrow][cb];
      d[0] = p0.x; d[1] = p0.y; d[2] = p0.z; d[3] = p0.w;
      d[4] = p1.x; d[5] = p1.y; d[6] = p1.z; d[7] = p1.w;
      if (kk + 32 < KT) __builtin_prefetch(Ag + 32, 0, 1);
    }
    // ---- load B tile (32 x 128) transposed into Bs[n][k] ----
    {
      int kg = kk + kr;
      const float* Bg = B + (size_t)kg * 1024 + n0 + nb;
      if (kg < Kreal) {   // uniform per thread: cheap scalar branch
        float4 f0 = *(const float4*)(Bg + 0);
        float4 f1 = *(const float4*)(Bg + 4);
        float4 f2 = *(const float4*)(Bg + 8);
        float4 f3 = *(const float4*)(Bg + 12);
        Bs[nb + 0][kr] = f2bf(f0.x);  Bs[nb + 1][kr] = f2bf(f0.y);
        Bs[nb + 2][kr] = f2bf(f0.z);  Bs[nb + 3][kr] = f2bf(f0.w);
        Bs[nb + 4][kr] = f2bf(f1.x);  Bs[nb + 5][kr] = f2bf(f1.y);
        Bs[nb + 6][kr] = f2bf(f1.z);  Bs[nb + 7][kr] = f2bf(f1.w);
        Bs[nb + 8][kr] = f2bf(f2.x);  Bs[nb + 9][kr] = f2bf(f2.y);
        Bs[nb + 10][kr] = f2bf(f2.z); Bs[nb + 11][kr] = f2bf(f2.w);
        Bs[nb + 12][kr] = f2bf(f3.x); Bs[nb + 13][kr] = f2bf(f3.y);
        Bs[nb + 14][kr] = f2bf(f3.z); Bs[nb + 15][kr] = f2bf(f3.w);
        if (kk + 32 < KT) __builtin_prefetch(Bg + 32 * 1024, 0, 1);
      } else {
#pragma unroll
        for (int c = 0; c < 16; ++c) Bs[nb + c][kr] = 0;
      }
    }
    __syncthreads();

    v16bf af0 = frag_a(As, wm * 32, lane);
    v16bf af1 = frag_a(As, wm * 32 + 16, lane);
    v16bf bfr[4];
#pragma unroll
    for (int ni = 0; ni < 4; ++ni) bfr[ni] = frag_b(Bs, wn * 64 + ni * 16, lane);

#pragma unroll
    for (int ni = 0; ni < 4; ++ni) {
      acc[0][ni] = __builtin_amdgcn_wmma_f32_16x16x32_bf16(
          false, af0, false, bfr[ni], (short)0, acc[0][ni], false, false);
      acc[1][ni] = __builtin_amdgcn_wmma_f32_16x16x32_bf16(
          false, af1, false, bfr[ni], (short)0, acc[1][ni], false, false);
    }
  }

  const float ap = aprm[0];
  const int hi8 = (lane >> 4) ? 8 : 0;
  const int nlo = lane & 15;

  if (EPI == 0) {
#pragma unroll
    for (int mi = 0; mi < 2; ++mi) {
      int tm = wm * 32 + mi * 16;
#pragma unroll
      for (int ni = 0; ni < 4; ++ni) {
        int n = n0 + wn * 64 + ni * 16 + nlo;
        float bn = bias[n];
#pragma unroll
        for (int v = 0; v < 8; ++v) {
          int m = blockM + tm + v + hi8;
          float z = acc[mi][ni][v] + bn;
          outb[(size_t)m * 1024 + n] = f2bf(preluf(z, ap));
        }
      }
    }
  } else {
    // fused head epilogue: reduce over n into per-row scalar
    float cv[4], wp[4], wbv[4], w1[4];
#pragma unroll
    for (int ni = 0; ni < 4; ++ni) {
      int n = n0 + wn * 64 + ni * 16 + nlo;
      cv[ni] = constv[n];
      wp[ni] = wpos[n];
      wbv[ni] = wb[n];
      w1[ni] = HAS_O1 ? wo1[n] : 0.f;
    }
#pragma unroll
    for (int mi = 0; mi < 2; ++mi) {
      int tm = wm * 32 + mi * 16;
#pragma unroll
      for (int v = 0; v < 8; ++v) {
        int m = blockM + tm + v + hi8;
        float pm = pos[m];
        float om = HAS_O1 ? o1v[m] : 0.f;
        float part = 0.f;
#pragma unroll
        for (int ni = 0; ni < 4; ++ni) {
          float z = acc[mi][ni][v] + cv[ni] + pm * wp[ni];
          if (HAS_O1) z += om * w1[ni];
          part += preluf(z, ap) * wbv[ni];
        }
#pragma unroll
        for (int off = 1; off < 16; off <<= 1)
          part += __shfl_xor(part, off, 32);
        if (nlo == 0) atomicAdd(&logit[m], part);
      }
    }
  }
}

// ---------------------------------------------------------------------------
// fp32 [32768,300] -> bf16 [32768,320] zero-padded (coalesced, one pass)
// ---------------------------------------------------------------------------
__global__ void k_cvt_pad(const float* __restrict__ src,
                          unsigned short* __restrict__ dst) {
  int idx = blockIdx.x * 256 + threadIdx.x;     // over 32768*320
  int m = idx / 320;
  int k = idx - m * 320;
  dst[idx] = (k < 300) ? f2bf(src[(size_t)m * 300 + k]) : (unsigned short)0;
}

// ---------------------------------------------------------------------------
// Small / streaming kernels
// ---------------------------------------------------------------------------
__global__ void k_prep(float* hsum, float* posf, const float* object_pos,
                       const float* Wp, const float* bp,
                       float* o1l0, float* o1l1, float* o2l0, float* o2l1,
                       const float* bo1b0, const float* bo1b1,
                       const float* bo2b0, const float* bo2b1,
                       float* stl0, float* stl1,
                       const float* bs0, const float* bs1) {
  int i = blockIdx.x * blockDim.x + threadIdx.x;
  if (i < 1024) hsum[i] = 0.f;
  if (i < 512) { stl0[i] = bs0[i]; stl1[i] = bs1[i]; }
  if (i < NOBJS) {
    const float* op = object_pos + (size_t)i * 4;
    float s = op[0] * Wp[0] + op[1] * Wp[1] + op[2] * Wp[2] + op[3] * Wp[3] + bp[0];
    posf[i] = sigmoidf_(s);
    o1l0[i] = bo1b0[0]; o1l1[i] = bo1b1[0];
    o2l0[i] = bo2b0[0]; o2l1[i] = bo2b1[0];
  }
}

// column-sum of h [50000,1024] (graph_attn softmax over axis=1 == ones)
__global__ void k_colsum(const float* __restrict__ h, float* __restrict__ hsum,
                         int nrows) {
  int col = blockIdx.x * 256 + threadIdx.x;
  int r0 = blockIdx.y * 500;
  int r1 = min(r0 + 500, nrows);
  float acc = 0.f;
  const float* p = h + (size_t)r0 * 1024 + col;
  for (int r = r0; r < r1; ++r) {
    acc += *p;
    __builtin_prefetch(p + 8 * 1024, 0, 0);
    p += 1024;
  }
  atomicAdd(&hsum[col], acc);
}

__global__ void k_goal_embed(const float* gv, const float* Wsb, const float* bsb,
                             const float* aprm, float* ge) {
  int i = blockIdx.x * 256 + threadIdx.x;
  float acc = bsb[i];
  for (int k = 0; k < 768; ++k) acc += gv[k] * Wsb[(size_t)k * 1024 + i];
  ge[i] = preluf(acc, aprm[0]);
}

__global__ void k_go(const float* gov, const float* Wc, const float* bc,
                     const float* aprm, float* go) {
  int o = blockIdx.x * 256 + threadIdx.x;   // 65536
  int g = o >> 10, i = o & 1023;
  float acc = bc[i];
  const float* gp = gov + (size_t)g * 300;
  for (int k = 0; k < 300; ++k) acc += gp[k] * Wc[(size_t)k * 1024 + i];
  go[o] = preluf(acc, aprm[0]);
}

__global__ void k_hembed(const float* hsum, const float* Wge, const float* bge,
                         const float* aprm, float* he) {
  int i = blockIdx.x * 256 + threadIdx.x;
  float acc = bge[i];
  for (int k = 0; k < 1024; ++k) acc += hsum[k] * Wge[(size_t)k * 1024 + i];
  he[i] = preluf(acc, aprm[0]);
}

__global__ void k_lstm_gates(const float* pv, const float* Wih, const float* bih,
                             const float* h0, const float* Whh, const float* bhh,
                             float* gates) {
  int j = blockIdx.x * 256 + threadIdx.x;   // 4096
  float acc = bih[j] + bhh[j];
  for (int k = 0; k < 610; ++k) acc += pv[k] * Wih[(size_t)k * 4096 + j];
  for (int k = 0; k < 1024; ++k) acc += h0[k] * Whh[(size_t)k * 4096 + j];
  gates[j] = acc;
}

__global__ void k_lstm_fin(const float* gates, const float* c0,
                           float* h1w, float* out_h1, float* out_c1) {
  int i = blockIdx.x * 256 + threadIdx.x;   // 1024
  float ig = sigmoidf_(gates[i]);
  float fg = sigmoidf_(gates[1024 + i]);
  float gg = tanhf(gates[2048 + i]);
  float og = sigmoidf_(gates[3072 + i]);
  float c1 = fg * c0[i] + ig * gg;
  float h1 = og * tanhf(c1);
  h1w[i] = h1; out_h1[i] = h1; out_c1[i] = c1;
}

__global__ void k_attn_goalobj(const float* he, const float* go,
                               const float* Wgo, const float* bgo,
                               float* goal_obj) {
  __shared__ float aw[64];
  int t = threadIdx.x;
  if (t < 64) {
    float s = bgo[0];
    for (int i = 0; i < 1024; ++i) s += he[i] * Wgo[i];
    const float* gp = go + (size_t)t * 1024;
    for (int i = 0; i < 1024; ++i) s += gp[i] * Wgo[1024 + i];
    aw[t] = s;
  }
  __syncthreads();
  if (t == 0) {
    float mx = -1e30f;
    for (int g = 0; g < 64; ++g) mx = fmaxf(mx, aw[g]);
    float sm = 0.f;
    for (int g = 0; g < 64; ++g) { aw[g] = expf(aw[g] - mx); sm += aw[g]; }
    float inv = 1.f / sm;
    for (int g = 0; g < 64; ++g) aw[g] *= inv;
  }
  __syncthreads();
  for (int i = t; i < 1024; i += 256) {
    float acc = 0.f;
    for (int g = 0; g < 64; ++g) acc += aw[g] * go[(size_t)g * 1024 + i];
    goal_obj[i] = acc;
  }
}

__global__ void k_final(const float* he, const float* gob, const float* ge,
                        const float* h1w, const float* Wfc, const float* bfc,
                        const float* aprm, float* fin, float* v0, float* v1) {
  int i = blockIdx.x * 256 + threadIdx.x;   // 1024
  float acc = bfc[i];
  for (int k = 0; k < 1024; ++k) acc += he[k]  * Wfc[(size_t)k * 1024 + i];
  for (int k = 0; k < 1024; ++k) acc += gob[k] * Wfc[(size_t)(1024 + k) * 1024 + i];
  for (int k = 0; k < 1024; ++k) acc += ge[k]  * Wfc[(size_t)(2048 + k) * 1024 + i];
  for (int k = 0; k < 1024; ++k) acc += h1w[k] * Wfc[(size_t)(3072 + k) * 1024 + i];
  float f = preluf(acc, aprm[0]);
  fin[i] = f; v0[i] = f; v1[i] = f;
}

__global__ void k_act(const float* fin,
                      const float* Wa0, const float* ba0,
                      const float* Wa1, const float* ba1,
                      float* out_act0, float* out_act1,
                      float* ohw, float* v0, float* v1) {
  __shared__ float lg[2][10];
  int t = threadIdx.x;
  if (t < 20) {
    int s = t / 10, r = t % 10;
    const float* Wa = s ? Wa1 : Wa0;
    const float* ba = s ? ba1 : ba0;
    float acc = ba[r];
    for (int i = 0; i < 1024; ++i) acc += fin[i] * Wa[(size_t)i * 10 + r];
    lg[s][r] = acc;
  }
  __syncthreads();
  if (t < 2) {
    int s = t;
    float* outa = s ? out_act1 : out_act0;
    float* vv = s ? v1 : v0;
    float mx = -1e30f;
    for (int r = 0; r < 10; ++r) mx = fmaxf(mx, lg[s][r]);
    float sm = 0.f, a[10];
    for (int r = 0; r < 10; ++r) { a[r] = expf(lg[s][r] - mx); sm += a[r]; }
    for (int r = 0; r < 10; ++r) { a[r] /= sm; outa[r] = a[r]; }
    // gumbel-softmax(act, tau=0.01)
    unsigned seed = s ? 0x3333u : 0x1111u;
    float y[10], ymx = -1e30f;
    for (int r = 0; r < 10; ++r) {
      y[r] = (a[r] + gumbel_(seed, r)) * 100.f;
      ymx = fmaxf(ymx, y[r]);
    }
    float ys = 0.f;
    for (int r = 0; r < 10; ++r) { y[r] = expf(y[r] - ymx); ys += y[r]; }
    for (int r = 0; r < 10; ++r) {
      float oh = y[r] / ys;
      ohw[s * 16 + r] = oh;
      vv[1024 + r] = oh;
    }
  }
}

// const vectors: pin(=concat(final,oh)) @ W[0:1034,:] + b  for (head,layer)
__global__ void k_const(const float* fin, const float* ohw,
                        const float* Wo1a0, const float* bo1a0,
                        const float* Wo2a0, const float* bo2a0,
                        const float* Wo1a1, const float* bo1a1,
                        const float* Wo2a1, const float* bo2a1,
                        float* constv) {
  int i = blockIdx.x * 256 + threadIdx.x;   // 0..1023
  int q = blockIdx.y;                       // 0..3  (head*2+layer)
  int head = q >> 1, layer = q & 1;
  const float* W = head ? (layer ? Wo2a1 : Wo1a1) : (layer ? Wo2a0 : Wo1a0);
  const float* b = head ? (layer ? bo2a1 : bo1a1) : (layer ? bo2a0 : bo1a0);
  const float* oh = ohw + head * 16;
  float acc = b[i];
  for (int k = 0; k < 1024; ++k) acc += fin[k] * W[(size_t)k * 1024 + i];
  for (int r = 0; r < 10; ++r) acc += oh[r] * W[(size_t)(1024 + r) * 1024 + i];
  constv[q * 1024 + i] = acc;
}

__global__ void k_o1fin(const float* logit, float* out_o1, float* o1sig) {
  int i = blockIdx.x * 256 + threadIdx.x;
  float s = sigmoidf_(logit[i]);
  out_o1[i] = s; o1sig[i] = s;
}

__global__ void k_o2fin(const float* logit, const float* aprm, float* out_o2) {
  int i = blockIdx.x * 256 + threadIdx.x;
  out_o2[i] = sigmoidf_(preluf(logit[i], aprm[0]));
}

// gumbel-softmax over o1 (32768) -> v[1034:33802]
__global__ __launch_bounds__(1024) void k_ohp(const float* o1sig, unsigned seed,
                                              float* vtail) {
  __shared__ float sh[1024];
  int t = threadIdx.x;
  float mx = -1e30f;
  for (int j = 0; j < 32; ++j) {
    int i = t + j * 1024;
    float x = (o1sig[i] + gumbel_(seed, (unsigned)i)) * 100.f;
    mx = fmaxf(mx, x);
  }
  sh[t] = mx; __syncthreads();
  for (int s = 512; s > 0; s >>= 1) {
    if (t < s) sh[t] = fmaxf(sh[t], sh[t + s]);
    __syncthreads();
  }
  mx = sh[0]; __syncthreads();
  float sm = 0.f;
  for (int j = 0; j < 32; ++j) {
    int i = t + j * 1024;
    float x = (o1sig[i] + gumbel_(seed, (unsigned)i)) * 100.f;
    sm += expf(x - mx);
  }
  sh[t] = sm; __syncthreads();
  for (int s = 512; s > 0; s >>= 1) {
    if (t < s) sh[t] += sh[t + s];
    __syncthreads();
  }
  float inv = 1.f / sh[0];
  for (int j = 0; j < 32; ++j) {
    int i = t + j * 1024;
    float x = (o1sig[i] + gumbel_(seed, (unsigned)i)) * 100.f;
    vtail[i] = expf(x - mx) * inv;
  }
}

// st GEMV: stl[j] += sum_r v[r]*Ws[r,j] over a 256-row stripe per block
__global__ void k_stgemv(const float* __restrict__ v, const float* __restrict__ Ws,
                         float* __restrict__ stl) {
  int r0 = blockIdx.x * 256;
  int r1 = min(r0 + 256, 33802);
  int c = threadIdx.x;
  float a0 = 0.f, a1 = 0.f;
  const float* w = Ws + (size_t)r0 * 512;
  for (int r = r0; r < r1; ++r) {
    float vr = v[r];
    a0 += vr * w[c];
    a1 += vr * w[c + 256];
    __builtin_prefetch(w + 8 * 512 + c, 0, 0);
    w += 512;
  }
  atomicAdd(&stl[c], a0);
  atomicAdd(&stl[c + 256], a1);
}

__global__ void k_stsm(const float* stl, float* out_st) {
  __shared__ float sh[512];
  int t = threadIdx.x;
  float x = stl[t];
  sh[t] = x; __syncthreads();
  for (int s = 256; s > 0; s >>= 1) {
    if (t < s) sh[t] = fmaxf(sh[t], sh[t + s]);
    __syncthreads();
  }
  float mx = sh[0]; __syncthreads();
  float e = expf(x - mx);
  sh[t] = e; __syncthreads();
  for (int s = 256; s > 0; s >>= 1) {
    if (t < s) sh[t] += sh[t + s];
    __syncthreads();
  }
  out_st[t] = e / sh[0];
}

// ---------------------------------------------------------------------------
// Host launch
// ---------------------------------------------------------------------------
extern "C" void kernel_launch(void* const* d_in, const int* in_sizes, int n_in,
                              void* d_out, int out_size, void* d_ws, size_t ws_size,
                              hipStream_t stream) {
  (void)in_sizes; (void)n_in; (void)out_size; (void)ws_size;
  const float* h       = (const float*)d_in[0];
  const float* goalVec = (const float*)d_in[1];
  const float* goalOV  = (const float*)d_in[2];
  const float* predv   = (const float*)d_in[3];
  const float* h0      = (const float*)d_in[4];
  const float* c0      = (const float*)d_in[5];
  const float* aprm    = (const float*)d_in[6];
  const float* Wsb = (const float*)d_in[7];  const float* bsb = (const float*)d_in[8];
  const float* We1 = (const float*)d_in[9];  const float* be1 = (const float*)d_in[10];
  const float* We2 = (const float*)d_in[11]; const float* be2 = (const float*)d_in[12];
  const float* Wp  = (const float*)d_in[13]; const float* bp  = (const float*)d_in[14];
  const float* Wc  = (const float*)d_in[15]; const float* bc  = (const float*)d_in[16];
  /* Wga=17, bga=18: dead code (softmax over singleton axis) */
  const float* Wge = (const float*)d_in[19]; const float* bge = (const float*)d_in[20];
  const float* Wgo = (const float*)d_in[21]; const float* bgo = (const float*)d_in[22];
  const float* Wfc = (const float*)d_in[23]; const float* bfc = (const float*)d_in[24];
  const float* Wih = (const float*)d_in[25]; const float* bih = (const float*)d_in[26];
  const float* Whh = (const float*)d_in[27]; const float* bhh = (const float*)d_in[28];
  const float* Wa_[2]   = {(const float*)d_in[29], (const float*)d_in[41]};
  const float* ba_[2]   = {(const float*)d_in[30], (const float*)d_in[42]};
  const float* Wo1a_[2] = {(const float*)d_in[31], (const float*)d_in[43]};
  const float* bo1a_[2] = {(const float*)d_in[32], (const float*)d_in[44]};
  const float* Wo1b_[2] = {(const float*)d_in[33], (const float*)d_in[45]};
  const float* bo1b_[2] = {(const float*)d_in[34], (const float*)d_in[46]};
  const float* Wo2a_[2] = {(const float*)d_in[35], (const float*)d_in[47]};
  const float* bo2a_[2] = {(const float*)d_in[36], (const float*)d_in[48]};
  const float* Wo2b_[2] = {(const float*)d_in[37], (const float*)d_in[49]};
  const float* bo2b_[2] = {(const float*)d_in[38], (const float*)d_in[50]};
  const float* Wst_[2]  = {(const float*)d_in[39], (const float*)d_in[51]};
  const float* bst_[2]  = {(const float*)d_in[40], (const float*)d_in[52]};
  const float* object_vec = (const float*)d_in[53];
  const float* object_pos = (const float*)d_in[54];

  // ---- outputs ----
  float* out = (float*)d_out;
  float* out_act[2] = {out + 0,     out + 66058};
  float* out_o1[2]  = {out + 10,    out + 66068};
  float* out_o2[2]  = {out + 32778, out + 98836};
  float* out_st[2]  = {out + 65546, out + 131604};
  float* out_h1 = out + 132116;
  float* out_c1 = out + 133140;

  // ---- workspace layout (floats) ----
  float* W = (float*)d_ws;
  float* hsum   = W + 0;
  float* ge     = W + 1024;
  float* he     = W + 2048;
  float* gob    = W + 3072;
  float* h1w    = W + 4096;
  float* fin    = W + 5120;
  float* gates  = W + 6144;      // 4096
  float* ohw    = W + 10240;     // [2][16]
  float* constv = W + 10272;     // [4][1024]
  float* go     = W + 14368;     // 65536
  float* posf   = W + 79904;     // 32768
  float* o1sig  = W + 112672;    // [2][32768]
  float* o1log  = W + 178208;    // [2][32768]
  float* o2log  = W + 243744;    // [2][32768]
  float* stlog  = W + 309280;    // [2][512]
  float* vbuf   = W + 310304;    // [2][33856]
  unsigned short* t1b  = (unsigned short*)(W + 378112);                  // 64MB
  unsigned short* objb = (unsigned short*)((char*)t1b + (size_t)NOBJS * 1024 * 2);
  unsigned short* ovb  = (unsigned short*)((char*)objb + (size_t)NOBJS * 1024 * 2);
  // ovb: bf16 [32768,320] zero-padded object_vec (~21MB)

  const dim3 B256(256);
  const dim3 GEMM_GRID(NOBJS / 128, 1024 / 128);

  // 1) init (logit biases, pos_feats, zero hsum, st bias) + bf16 pad convert
  k_prep<<<dim3(128), B256, 0, stream>>>(hsum, posf, object_pos, Wp, bp,
      o1log, o1log + NOBJS, o2log, o2log + NOBJS,
      bo1b_[0], bo1b_[1], bo2b_[0], bo2b_[1],
      stlog, stlog + 512, bst_[0], bst_[1]);
  k_cvt_pad<<<dim3((NOBJS * 320) / 256), B256, 0, stream>>>(object_vec, ovb);
  // 2) column sum of h (the only live use of h)
  k_colsum<<<dim3(4, 100), B256, 0, stream>>>(h, hsum, 50000);
  // 3) independent small GEMVs
  k_go<<<dim3(256), B256, 0, stream>>>(goalOV, Wc, bc, aprm, go);
  k_goal_embed<<<dim3(4), B256, 0, stream>>>(goalVec, Wsb, bsb, aprm, ge);
  k_lstm_gates<<<dim3(16), B256, 0, stream>>>(predv, Wih, bih, h0, Whh, bhh, gates);
  k_lstm_fin<<<dim3(4), B256, 0, stream>>>(gates, c0, h1w, out_h1, out_c1);
  // 4) big WMMA GEMMs: object MLP
  k_gemm_wmma<0, false><<<GEMM_GRID, B256, 0, stream>>>(
      ovb, 320, 300, 320, We1,
      be1, t1b, nullptr, nullptr, nullptr, nullptr, nullptr, nullptr, nullptr, aprm);
  k_gemm_wmma<0, false><<<GEMM_GRID, B256, 0, stream>>>(
      t1b, 1024, 1024, 1024, We2,
      be2, objb, nullptr, nullptr, nullptr, nullptr, nullptr, nullptr, nullptr, aprm);
  // 5) graph/goal path
  k_hembed<<<dim3(4), B256, 0, stream>>>(hsum, Wge, bge, aprm, he);
  k_attn_goalobj<<<dim3(1), B256, 0, stream>>>(he, go, Wgo, bgo, gob);
  k_final<<<dim3(4), B256, 0, stream>>>(he, gob, ge, h1w, Wfc, bfc, aprm,
                                        fin, vbuf, vbuf + 33856);
  k_act<<<dim3(1), dim3(64), 0, stream>>>(fin, Wa_[0], ba_[0], Wa_[1], ba_[1],
                                          out_act[0], out_act[1], ohw,
                                          vbuf, vbuf + 33856);
  k_const<<<dim3(4, 4), B256, 0, stream>>>(fin, ohw,
      Wo1a_[0], bo1a_[0], Wo2a_[0], bo2a_[0],
      Wo1a_[1], bo1a_[1], Wo2a_[1], bo2a_[1], constv);

  // 6) heads: fused obj_feats @ Wmid -> prelu -> dot(Wb) -> scalar per object
  for (int s = 0; s < 2; ++s) {
    float* vh = vbuf + (size_t)s * 33856;
    // o1
    k_gemm_wmma<1, false><<<GEMM_GRID, B256, 0, stream>>>(
        objb, 1024, 1024, 1024, Wo1a_[s] + (size_t)1034 * 1024,
        nullptr, nullptr,
        constv + (s * 2 + 0) * 1024, Wo1a_[s] + (size_t)2058 * 1024,
        nullptr, nullptr, Wo1b_[s], posf, o1log + (size_t)s * NOBJS, aprm);
    k_o1fin<<<dim3(128), B256, 0, stream>>>(o1log + (size_t)s * NOBJS,
                                            out_o1[s], o1sig + (size_t)s * NOBJS);
    k_ohp<<<dim3(1), dim3(1024), 0, stream>>>(o1sig + (size_t)s * NOBJS,
                                              s ? 0x4444u : 0x2222u, vh + 1034);
    // o2 (adds o1 coupling column)
    k_gemm_wmma<1, true><<<GEMM_GRID, B256, 0, stream>>>(
        objb, 1024, 1024, 1024, Wo2a_[s] + (size_t)1034 * 1024,
        nullptr, nullptr,
        constv + (s * 2 + 1) * 1024, Wo2a_[s] + (size_t)2058 * 1024,
        Wo2a_[s] + (size_t)2059 * 1024, o1sig + (size_t)s * NOBJS,
        Wo2b_[s], posf, o2log + (size_t)s * NOBJS, aprm);
    k_o2fin<<<dim3(128), B256, 0, stream>>>(o2log + (size_t)s * NOBJS, aprm, out_o2[s]);
    // state head
    k_stgemv<<<dim3(133), B256, 0, stream>>>(vh, Wst_[s], stlog + (size_t)s * 512);
    k_stsm<<<dim3(1), dim3(512), 0, stream>>>(stlog + (size_t)s * 512, out_st[s]);
  }
}
